// PoseAttnNet_83786222011106
// MI455X (gfx1250) — matrix operand order repaired and checked by reading, not compiled, not run on hardware
//
#include <hip/hip_runtime.h>
#include <hip/hip_bf16.h>
#include <math.h>

// ---------------- problem constants ----------------
#define B_     4
#define N_     2048
#define NT_    (N_ / 16)   // 128 16-wide tiles per sequence
#define DIM_   384
#define H_     8
#define HD_    48          // head dim
#define HDP_   64          // padded head dim (multiple of 32 for WMMA K)
#define POSED_ 128
#define HID_   1536
#define M_     (B_ * N_)   // 8192 token rows

typedef _Float16 h16;
typedef __attribute__((ext_vector_type(16))) _Float16 v16h;
typedef __attribute__((ext_vector_type(8)))  _Float16 v8h;
typedef __attribute__((ext_vector_type(8)))  float    v8f;
typedef __attribute__((ext_vector_type(4)))  int      v4i;

#define AS1 __attribute__((address_space(1)))
#define AS3 __attribute__((address_space(3)))

#if __has_builtin(__builtin_amdgcn_global_load_async_to_lds_b128)
#define HAVE_ASYNC_LDS 1
#else
#define HAVE_ASYNC_LDS 0
#endif

// ---------------- WMMA helper ----------------
__device__ __forceinline__ v8f wmma_f16(v16h a, v16h b, v8f c) {
    return __builtin_amdgcn_wmma_f32_16x16x32_f16(
        false, a, false, b, (short)0, c, false, false);
}

// ---------------- async global -> LDS staging (16B granule) ----------------
__device__ __forceinline__ void cp16_g2l(const h16* g, h16* s) {
#if HAVE_ASYNC_LDS
    __builtin_amdgcn_global_load_async_to_lds_b128(
        (AS1 v4i*)g, (AS3 v4i*)s, 0, 0);
#else
    *(v8h*)s = *(const v8h*)g;
#endif
}
__device__ __forceinline__ void wait_async_all() {
#if HAVE_ASYNC_LDS
#if __has_builtin(__builtin_amdgcn_s_wait_asynccnt)
    __builtin_amdgcn_s_wait_asynccnt(0);
#else
    asm volatile("s_wait_asynccnt 0" ::: "memory");
#endif
#endif
}

// ---------------- DPP reductions across each 16-lane half ----------------
template <int CTRL>
__device__ __forceinline__ float dpp_f(float v) {
    return __builtin_bit_cast(float,
        __builtin_amdgcn_update_dpp(0, __builtin_bit_cast(int, v),
                                    CTRL, 0xf, 0xf, true));
}
__device__ __forceinline__ float half16_max(float x) {
    x = fmaxf(x, dpp_f<0xB1>(x));   // quad_perm xor1
    x = fmaxf(x, dpp_f<0x4E>(x));   // quad_perm xor2
    x = fmaxf(x, dpp_f<0x141>(x));  // row_half_mirror
    x = fmaxf(x, dpp_f<0x140>(x));  // row_mirror
    return x;
}
__device__ __forceinline__ float half16_sum(float x) {
    x += dpp_f<0xB1>(x);
    x += dpp_f<0x4E>(x);
    x += dpp_f<0x141>(x);
    x += dpp_f<0x140>(x);
    return x;
}

// A-fragment: 16x32 f16 tile, row-major source, lda = K-stride in elements.
__device__ __forceinline__ v16h load_a_frag(const h16* __restrict__ base, int lda) {
    int lane    = threadIdx.x & 31;
    int half_id = lane >> 4;
    int m       = lane & 15;
    const h16* p = base + (long)m * lda + half_id * 8;
    v8h c0 = *(const v8h*)(p);
    v8h c1 = *(const v8h*)(p + 16);
    v16h a;
#pragma unroll
    for (int i = 0; i < 8; ++i) { a[i] = c0[i]; a[i + 8] = c1[i]; }
    return a;
}

// B-fragment: 32x16 f16 tile from K-contiguous layout Bt[n][k], ldb = K-stride.
__device__ __forceinline__ v16h load_b_frag(const h16* __restrict__ base, int ldb) {
    int lane    = threadIdx.x & 31;
    int half_id = lane >> 4;
    int n       = lane & 15;
    const h16* p = base + (long)n * ldb + half_id * 8;
    v8h c0 = *(const v8h*)(p);
    v8h c1 = *(const v8h*)(p + 16);
    v16h b;
#pragma unroll
    for (int i = 0; i < 8; ++i) { b[i] = c0[i]; b[i + 8] = c1[i]; }
    return b;
}

__device__ __forceinline__ float gelu_exact(float x) {
    return 0.5f * x * (1.0f + erff(x * 0.7071067811865476f));
}

// ---------------- weight transpose + f32->f16 convert ----------------
__global__ __launch_bounds__(256) void wconv_kernel(const float* __restrict__ src,
                                                    h16* __restrict__ dst,
                                                    int K, int Nc) {
    long i = (long)blockIdx.x * 256 + threadIdx.x;
    if (i >= (long)K * Nc) return;
    int k = (int)(i / Nc);
    int n = (int)(i % Nc);
    dst[(long)n * K + k] = (h16)src[i];
}

__global__ __launch_bounds__(256) void sconv_kernel(const float* __restrict__ src,
                                                    h16* __restrict__ dst,
                                                    long count, float s) {
    long i = (long)blockIdx.x * 256 + threadIdx.x;
    if (i < count) dst[i] = (h16)(src[i] * s);
}

// ---------------- LayerNorm: one wave per 384-wide row ----------------
__global__ __launch_bounds__(256) void ln_kernel(const float* __restrict__ x,
                                                 const float* __restrict__ g,
                                                 const float* __restrict__ b,
                                                 h16* __restrict__ out) {
    int wave = threadIdx.x >> 5;
    int lane = threadIdx.x & 31;
    long row = (long)blockIdx.x * 8 + wave;
    const float* xr = x + row * DIM_;
    float v[12];
    float sum = 0.0f;
#pragma unroll
    for (int i = 0; i < 12; ++i) { v[i] = xr[lane + 32 * i]; sum += v[i]; }
#pragma unroll
    for (int m = 16; m >= 1; m >>= 1) sum += __shfl_xor(sum, m, 32);
    float mu = sum * (1.0f / DIM_);
    float var = 0.0f;
#pragma unroll
    for (int i = 0; i < 12; ++i) { float d = v[i] - mu; var += d * d; }
#pragma unroll
    for (int m = 16; m >= 1; m >>= 1) var += __shfl_xor(var, m, 32);
    var *= (1.0f / DIM_);
    float inv = rsqrtf(var + 1e-5f);
    h16* orow = out + row * DIM_;
#pragma unroll
    for (int i = 0; i < 12; ++i) {
        int c = lane + 32 * i;
        orow[c] = (h16)((v[i] - mu) * inv * g[c] + b[c]);
    }
}

// ---------------- WMMA GEMM core, async-LDS-staged B, epilogues ----------------
// EPI 0: f16 out, optional bias, scale (or sqrt(*scaleptr/64))
// EPI 1: f16 out, bias + exact GELU
// EPI 2: f16 scatter into [B,H,N,HDP] (qh/kh), * scale
// EPI 3: f16 scatter into [B,H,HD,N]  (v transposed)
// EPI 4: f32 out = acc + bias + resid
// EPI 5: f16 tiled fragment-order out (pose bias): [bz][qt][jt][lane][8]
template <int EPI>
__global__ __launch_bounds__(128) void gemm_kernel(
    const h16* __restrict__ A, const h16* __restrict__ Bt,
    const float* __restrict__ bias, void* __restrict__ out,
    const float* __restrict__ resid, const float* __restrict__ scaleptr,
    int M, int K, int Nout, int lda, int ldb,
    long strideA, long strideB, long strideC, float scale) {

    alignas(16) __shared__ h16 sB[2][64 * 32];   // double-buffered 64col x 32k tile

    int tid  = threadIdx.x;
    int wave = tid >> 5;
    int lane = tid & 31;
    int row0 = (blockIdx.x * 4 + wave) * 16;
    int col0 = blockIdx.y * 64;
    int bz   = blockIdx.z;

    const h16* Ab  = A  + (long)bz * strideA;
    const h16* Btb = Bt + (long)bz * strideB;

    // staging coords: thread covers 32B (16 halfs) of the 64x32 tile
    int sc_col  = tid >> 1;            // 0..63
    int sc_off  = (tid & 1) * 16;      // half-index offset within the 32-k row
    const h16* gB = Btb + (long)(col0 + sc_col) * ldb + sc_off;
    h16* sdst0 = &sB[0][sc_col * 32 + sc_off];
    h16* sdst1 = &sB[1][sc_col * 32 + sc_off];

    // prologue: stage k0 = 0 into buffer 0
    cp16_g2l(gB, sdst0);
    cp16_g2l(gB + 8, sdst0 + 8);

    v8f acc[4] = {v8f{}, v8f{}, v8f{}, v8f{}};
    int nk = K / 32;
    for (int i = 0; i < nk; ++i) {
        int k0 = i * 32;
        wait_async_all();
        __syncthreads();
        int cur = i & 1;
        if (i + 1 < nk) {
            h16* nd = (cur == 0) ? sdst1 : sdst0;
            const h16* ng = gB + (k0 + 32);
            cp16_g2l(ng, nd);
            cp16_g2l(ng + 8, nd + 8);
        }
        v16h af = load_a_frag(Ab + (long)row0 * lda + k0, lda);
#pragma unroll
        for (int t = 0; t < 4; ++t) {
            v16h bf = load_b_frag(&sB[cur][t * 16 * 32], 32);
            acc[t] = wmma_f16(af, bf, acc[t]);
        }
    }

    int half_id = lane >> 4;
    int nl      = lane & 15;

    if (EPI == 5) {
        // fragment-order tiles: [(bz*NT_ + qt)*NT_ + jt] * 256 + lane*8
        long tbase = ((long)bz * NT_ + (row0 >> 4)) * NT_ + (col0 >> 4);
#pragma unroll
        for (int t = 0; t < 4; ++t) {
            v8h ph;
#pragma unroll
            for (int r = 0; r < 8; ++r) ph[r] = (h16)acc[t][r];
            *(v8h*)((h16*)out + (tbase + t) * 256 + lane * 8) = ph;
        }
        return;
    }

    float sc = scale;
    if (EPI == 0 && scaleptr) sc = sqrtf(fmaxf(scaleptr[0], 0.0f) * (1.0f / 64.0f));

#pragma unroll
    for (int t = 0; t < 4; ++t) {
        int col = col0 + t * 16 + nl;
        float bv = (EPI == 0) ? (bias ? bias[col] : 0.0f)
                 : (EPI == 1 || EPI == 4) ? bias[col] : 0.0f;
#pragma unroll
        for (int r = 0; r < 8; ++r) {
            int  m   = r + 8 * half_id;
            long row = row0 + m;
            float v  = acc[t][r];
            if (EPI == 0) {
                h16* op = (h16*)out + (long)bz * strideC;
                op[row * Nout + col] = (h16)((v + bv) * sc);
            } else if (EPI == 1) {
                ((h16*)out)[row * Nout + col] = (h16)gelu_exact(v + bv);
            } else if (EPI == 2) {
                int b_tok = (int)(row >> 11);
                int n_tok = (int)(row & (N_ - 1));
                int h = col / HD_, d = col % HD_;
                ((h16*)out)[(((long)(b_tok * H_ + h) * N_ + n_tok) << 6) + d] = (h16)(v * sc);
            } else if (EPI == 3) {
                int b_tok = (int)(row >> 11);
                int n_tok = (int)(row & (N_ - 1));
                int h = col / HD_, d = col % HD_;
                ((h16*)out)[((long)(b_tok * H_ + h) * HD_ + d) * N_ + n_tok] = (h16)v;
            } else { // EPI == 4
                ((float*)out)[row * Nout + col] = v + bv + resid[row * Nout + col];
            }
        }
    }
}

// ---------------- flash attention with tiled pose bias ----------------
__global__ __launch_bounds__(32) void attn_kernel(
    const h16* __restrict__ qh,    // [B,H,N,HDP] (pre-scaled, pad zero)
    const h16* __restrict__ kh,    // [B,H,N,HDP] (pad zero)
    const h16* __restrict__ vt,    // [B,H,HD,N]
    const h16* __restrict__ pbias, // tiled: [B][qt][jt][lane][8] f16
    h16* __restrict__ o)           // [B,N,DIM]
{
    alignas(16) __shared__ h16 sP[16 * 32];
    int lane    = threadIdx.x & 31;
    int half_id = lane >> 4;
    int nl      = lane & 15;
    int q0 = blockIdx.x * 16;
    int h  = blockIdx.y;
    int b  = blockIdx.z;

    const h16* Q   = qh + ((long)(b * H_ + h) * N_ + q0) * HDP_;
    const h16* Kb  = kh + (long)(b * H_ + h) * N_ * HDP_;
    const h16* Vt  = vt + (long)(b * H_ + h) * HD_ * N_;
    const h16* PBt = pbias + (((long)b * NT_ + (q0 >> 4)) * NT_) * 256 + lane * 8;

    v16h aq0 = load_a_frag(Q, HDP_);
    v16h aq1 = load_a_frag(Q + 32, HDP_);

    v8f  o_acc[3] = {v8f{}, v8f{}, v8f{}};
    float mstat[8], lstat[8];
#pragma unroll
    for (int r = 0; r < 8; ++r) { mstat[r] = -1e30f; lstat[r] = 0.0f; }

    for (int j0 = 0; j0 < N_; j0 += 32) {
        // ---- S = Q K^T : 16 x 32 (two 16-col tiles)
        v8f s[2] = {v8f{}, v8f{}};
#pragma unroll
        for (int t = 0; t < 2; ++t) {
            v16h bk0 = load_b_frag(Kb + (long)(j0 + t * 16) * HDP_ + 0,  HDP_);
            s[t] = wmma_f16(aq0, bk0, s[t]);
            v16h bk1 = load_b_frag(Kb + (long)(j0 + t * 16) * HDP_ + 32, HDP_);
            s[t] = wmma_f16(aq1, bk1, s[t]);
        }
        // ---- add pose bias: fragment-layout tiles, one 16B load per tile
#pragma unroll
        for (int t = 0; t < 2; ++t) {
            v8h pb = *(const v8h*)(PBt + (long)((j0 >> 4) + t) * 256);
#pragma unroll
            for (int r = 0; r < 8; ++r) s[t][r] += (float)pb[r];
        }

        // ---- online softmax (DPP reductions across each 16-lane half)
        float alpha[8];
#pragma unroll
        for (int r = 0; r < 8; ++r) {
            float rmax = half16_max(fmaxf(s[0][r], s[1][r]));
            float mn = fmaxf(mstat[r], rmax);
            alpha[r] = __expf(mstat[r] - mn);
            mstat[r] = mn;
        }
#pragma unroll
        for (int t = 0; t < 2; ++t)
#pragma unroll
            for (int r = 0; r < 8; ++r)
                s[t][r] = __expf(s[t][r] - mstat[r]);
#pragma unroll
        for (int r = 0; r < 8; ++r) {
            float rsum = half16_sum(s[0][r] + s[1][r]);
            lstat[r] = lstat[r] * alpha[r] + rsum;
        }
#pragma unroll
        for (int c = 0; c < 3; ++c)
#pragma unroll
            for (int r = 0; r < 8; ++r)
                o_acc[c][r] *= alpha[r];

        // ---- transpose P (C layout -> A layout) through LDS
#pragma unroll
        for (int t = 0; t < 2; ++t)
#pragma unroll
            for (int r = 0; r < 8; ++r)
                sP[(r + 8 * half_id) * 32 + t * 16 + nl] = (h16)s[t][r];
        __syncthreads();
        v16h ap = load_a_frag(sP, 32);
#pragma unroll
        for (int c = 0; c < 3; ++c) {
            v16h bvf = load_b_frag(Vt + (long)(c * 16) * N_ + j0, N_);
            o_acc[c] = wmma_f16(ap, bvf, o_acc[c]);
        }
        __syncthreads();
    }

#pragma unroll
    for (int c = 0; c < 3; ++c)
#pragma unroll
        for (int r = 0; r < 8; ++r) {
            int  m  = r + 8 * half_id;
            float v = o_acc[c][r] / lstat[r];
            o[((long)b * N_ + q0 + m) * DIM_ + h * HD_ + c * 16 + nl] = (h16)v;
        }
}

// ---------------- host side ----------------
extern "C" void kernel_launch(void* const* d_in, const int* in_sizes, int n_in,
                              void* d_out, int out_size, void* d_ws, size_t ws_size,
                              hipStream_t stream) {
    (void)in_sizes; (void)n_in; (void)out_size; (void)ws_size;
    const float* q        = (const float*)d_in[0];
    const float* pose     = (const float*)d_in[1];
    const float* ln1_g    = (const float*)d_in[2];
    const float* ln1_b    = (const float*)d_in[3];
    const float* wq       = (const float*)d_in[4];
    const float* wk       = (const float*)d_in[5];
    const float* wv       = (const float*)d_in[6];
    const float* pose_w   = (const float*)d_in[7];
    const float* pose_b   = (const float*)d_in[8];
    const float* pose_wt  = (const float*)d_in[9];
    const float* proj_w   = (const float*)d_in[10];
    const float* proj_b   = (const float*)d_in[11];
    const float* ln2_g    = (const float*)d_in[12];
    const float* ln2_b    = (const float*)d_in[13];
    const float* fc1_w    = (const float*)d_in[14];
    const float* fc1_b    = (const float*)d_in[15];
    const float* fc2_w    = (const float*)d_in[16];
    const float* fc2_b    = (const float*)d_in[17];
    float* outp = (float*)d_out;

    char* ws = (char*)d_ws;
    size_t off = 0;
    auto alloc = [&](size_t bytes) -> void* {
        void* p = ws + off;
        off += (bytes + 255) & ~(size_t)255;
        return p;
    };

    h16* x16      = (h16*)alloc((size_t)M_ * DIM_ * 2);
    h16* pose16   = (h16*)alloc((size_t)M_ * POSED_ * 2);
    h16* pe16     = (h16*)alloc((size_t)M_ * DIM_ * 2);
    h16* wq16t    = (h16*)alloc((size_t)DIM_ * DIM_ * 2);
    h16* wk16t    = (h16*)alloc((size_t)DIM_ * DIM_ * 2);
    h16* wv16t    = (h16*)alloc((size_t)DIM_ * DIM_ * 2);
    h16* posew16t = (h16*)alloc((size_t)DIM_ * POSED_ * 2);
    h16* projw16t = (h16*)alloc((size_t)DIM_ * DIM_ * 2);
    h16* fc1w16t  = (h16*)alloc((size_t)DIM_ * HID_ * 2);
    h16* fc2w16t  = (h16*)alloc((size_t)DIM_ * HID_ * 2);
    h16* qh       = (h16*)alloc((size_t)B_ * H_ * N_ * HDP_ * 2);
    h16* kh       = (h16*)alloc((size_t)B_ * H_ * N_ * HDP_ * 2);
    h16* vt       = (h16*)alloc((size_t)B_ * H_ * HD_ * N_ * 2);
    h16* pbias    = (h16*)alloc((size_t)B_ * N_ * N_ * 2);   // tiled layout
    h16* attn_o   = (h16*)alloc((size_t)M_ * DIM_ * 2);
    float* q2     = (float*)alloc((size_t)M_ * DIM_ * 4);
    h16* y16      = (h16*)alloc((size_t)M_ * DIM_ * 2);
    h16* h1       = (h16*)alloc((size_t)M_ * HID_ * 2);

    auto wconv = [&](const float* s, h16* d, int K, int Nc) {
        long tot = (long)K * Nc;
        wconv_kernel<<<(unsigned)((tot + 255) / 256), 256, 0, stream>>>(s, d, K, Nc);
    };
    wconv(wq, wq16t, DIM_, DIM_);
    wconv(wk, wk16t, DIM_, DIM_);
    wconv(wv, wv16t, DIM_, DIM_);
    wconv(pose_w, posew16t, POSED_, DIM_);
    wconv(proj_w, projw16t, DIM_, DIM_);
    wconv(fc1_w, fc1w16t, DIM_, HID_);
    wconv(fc2_w, fc2w16t, HID_, DIM_);

    ln_kernel<<<M_ / 8, 256, 0, stream>>>(q, ln1_g, ln1_b, x16);
    {
        long cnt = (long)M_ * POSED_;
        sconv_kernel<<<(unsigned)((cnt + 255) / 256), 256, 0, stream>>>(pose, pose16, cnt, 0.01f);
    }

    (void)hipMemsetAsync(qh, 0, (size_t)B_ * H_ * N_ * HDP_ * 2, stream);
    (void)hipMemsetAsync(kh, 0, (size_t)B_ * H_ * N_ * HDP_ * 2, stream);

    const float qscale = 0.14433756729740643f; // 48^-0.5

    gemm_kernel<2><<<dim3(M_ / 64, DIM_ / 64, 1), 128, 0, stream>>>(
        x16, wq16t, nullptr, qh, nullptr, nullptr, M_, DIM_, DIM_, DIM_, DIM_, 0, 0, 0, qscale);
    gemm_kernel<2><<<dim3(M_ / 64, DIM_ / 64, 1), 128, 0, stream>>>(
        x16, wk16t, nullptr, kh, nullptr, nullptr, M_, DIM_, DIM_, DIM_, DIM_, 0, 0, 0, 1.0f);
    gemm_kernel<3><<<dim3(M_ / 64, DIM_ / 64, 1), 128, 0, stream>>>(
        x16, wv16t, nullptr, vt, nullptr, nullptr, M_, DIM_, DIM_, DIM_, DIM_, 0, 0, 0, 1.0f);

    gemm_kernel<0><<<dim3(M_ / 64, DIM_ / 64, 1), 128, 0, stream>>>(
        pose16, posew16t, pose_b, pe16, nullptr, pose_wt, M_, POSED_, DIM_, POSED_, POSED_, 0, 0, 0, 1.0f);

    // pose bias in fragment-tiled layout
    gemm_kernel<5><<<dim3(N_ / 64, N_ / 64, B_), 128, 0, stream>>>(
        pe16, pe16, nullptr, pbias, nullptr, nullptr, N_, DIM_, N_, DIM_, DIM_,
        (long)N_ * DIM_, (long)N_ * DIM_, 0, 1.0f);

    attn_kernel<<<dim3(N_ / 16, H_, B_), 32, 0, stream>>>(qh, kh, vt, pbias, attn_o);

    gemm_kernel<4><<<dim3(M_ / 64, DIM_ / 64, 1), 128, 0, stream>>>(
        attn_o, projw16t, proj_b, q2, q, nullptr, M_, DIM_, DIM_, DIM_, DIM_, 0, 0, 0, 1.0f);

    ln_kernel<<<M_ / 8, 256, 0, stream>>>(q2, ln2_g, ln2_b, y16);

    gemm_kernel<1><<<dim3(M_ / 64, HID_ / 64, 1), 128, 0, stream>>>(
        y16, fc1w16t, fc1_b, h1, nullptr, nullptr, M_, DIM_, HID_, DIM_, DIM_, 0, 0, 0, 1.0f);
    gemm_kernel<4><<<dim3(M_ / 64, DIM_ / 64, 1), 128, 0, stream>>>(
        h1, fc2w16t, fc2_b, outp, q2, nullptr, M_, HID_, DIM_, HID_, HID_, 0, 0, 0, 1.0f);
}
